// MultiHeadAttention_9560597201353
// MI455X (gfx1250) — matrix operand order repaired and checked by reading, not compiled, not run on hardware
//
#include <hip/hip_runtime.h>

#define B_DIM 4
#define S_DIM 2048
#define D_MODEL 1024
#define NUM_HEADS 16
#define D_K 64
#define M_ROWS (B_DIM * S_DIM)   // 8192
#define NEG_BIG (-1e9f)
#define SM_SCALE 0.125f          // 1/sqrt(64)

typedef __bf16 bf16_t;
typedef __attribute__((ext_vector_type(16))) __bf16 v16bf;
typedef __attribute__((ext_vector_type(8)))  __bf16 v8bf;
typedef __attribute__((ext_vector_type(8)))  float  v8f;

union FragBF {
    v16bf  v;
    v8bf   h[2];
    bf16_t e[16];
};

__device__ __forceinline__ v8f wmma_bf16(const FragBF& a, const FragBF& b, v8f c) {
    return __builtin_amdgcn_wmma_f32_16x16x32_bf16(false, a.v, false, b.v,
                                                   (short)0, c, false, false);
}

// A fragment (16x32), bf16 row-major source.
// lane<16 holds K=[k0..k0+7],[k0+16..k0+23]; lane>=16 holds K=[k0+8..15],[k0+24..31]
__device__ __forceinline__ void load_a_bf16(FragBF& f, const bf16_t* base, int ld,
                                            int row0, int k0, int lane) {
    int row = row0 + (lane & 15);
    int hh  = lane >> 4;
    const bf16_t* p = base + (size_t)row * ld + k0 + hh * 8;
    f.h[0] = *(const v8bf*)(p);
    f.h[1] = *(const v8bf*)(p + 16);
}

// A fragment from fp32 row-major source, converting to bf16 on the fly.
__device__ __forceinline__ void load_a_f32(FragBF& f, const float* base, int ld,
                                           int row0, int k0, int lane) {
    int row = row0 + (lane & 15);
    int hh  = lane >> 4;
    const float* p = base + (size_t)row * ld + k0 + hh * 8;
    float t[16];
    *(float4*)(t + 0)  = *(const float4*)(p + 0);
    *(float4*)(t + 4)  = *(const float4*)(p + 4);
    *(float4*)(t + 8)  = *(const float4*)(p + 16);
    *(float4*)(t + 12) = *(const float4*)(p + 20);
#pragma unroll
    for (int i = 0; i < 16; ++i) f.e[i] = (bf16_t)t[i];
}

// B fragment (32x16 KxN), source row-major [N][K] bf16 (== column-major KxN).
// lane<16 holds K=[k0..k0+7],[k0+16..k0+23]; lane>=16 holds K=[k0+8..15],[k0+24..31]
__device__ __forceinline__ void load_b_bf16(FragBF& f, const bf16_t* base, int ld,
                                            int n0, int k0, int lane) {
    const bf16_t* p = base + (size_t)(n0 + (lane & 15)) * ld + k0 + (lane >> 4) * 8;
    f.h[0] = *(const v8bf*)(p);
    f.h[1] = *(const v8bf*)(p + 16);
}

// ---------------------------------------------------------------- cvt kernel
__global__ void cvt_f32_bf16(const float* __restrict__ src, bf16_t* __restrict__ dst, int n) {
    int i = blockIdx.x * blockDim.x + threadIdx.x;
    for (; i < n; i += gridDim.x * blockDim.x) dst[i] = (bf16_t)src[i];
}

// ------------------------------------------------------------ projection GEMM
// y = A(fp32, M x K row-major) @ W(bf16, N x K row-major)^T + bias
// VT==0 : store to [b, h, s, d]   (q / k layout)
// VT==1 : store to [b, h, d, s]   (v transposed for P.V B-fragments)
template <int VT>
__global__ __launch_bounds__(256) void proj_kernel(const float* __restrict__ A,
                                                   const bf16_t* __restrict__ W,
                                                   const float* __restrict__ bias,
                                                   bf16_t* __restrict__ out) {
    const int lane = threadIdx.x & 31;
    const int wid  = threadIdx.x >> 5;
    const int wm = wid & 3, wn = wid >> 2;          // 4 x 2 wave grid
    const int gx = blockIdx.x & 7;                  // N / 128 = 8
    const int gy = blockIdx.x >> 3;                 // M / 128 = 64
    const int m0 = gy * 128 + wm * 32;
    const int n0 = gx * 128 + wn * 64;

    v8f acc[2][4];
#pragma unroll
    for (int ms = 0; ms < 2; ++ms)
#pragma unroll
        for (int ns = 0; ns < 4; ++ns)
            acc[ms][ns] = (v8f){0.f, 0.f, 0.f, 0.f, 0.f, 0.f, 0.f, 0.f};

    for (int k0 = 0; k0 < D_MODEL; k0 += 32) {
        FragBF a[2];
        load_a_f32(a[0], A, D_MODEL, m0,      k0, lane);
        load_a_f32(a[1], A, D_MODEL, m0 + 16, k0, lane);
        FragBF b[4];
#pragma unroll
        for (int ns = 0; ns < 4; ++ns)
            load_b_bf16(b[ns], W, D_MODEL, n0 + ns * 16, k0, lane);
#pragma unroll
        for (int ms = 0; ms < 2; ++ms)
#pragma unroll
            for (int ns = 0; ns < 4; ++ns)
                acc[ms][ns] = wmma_bf16(a[ms], b[ns], acc[ms][ns]);
    }

    const int hh = lane >> 4;
    const int cn = lane & 15;
#pragma unroll
    for (int ms = 0; ms < 2; ++ms) {
#pragma unroll
        for (int ns = 0; ns < 4; ++ns) {
            const int n  = n0 + ns * 16 + cn;
            const float bv = bias[n];
            const int hId = n >> 6, d = n & 63;
#pragma unroll
            for (int i = 0; i < 8; ++i) {
                const int row = m0 + ms * 16 + i + 8 * hh;
                const int bb = row >> 11, s = row & 2047;
                const float val = acc[ms][ns][i] + bv;
                if (VT)
                    out[(((size_t)(bb * NUM_HEADS + hId)) * D_K + d) * S_DIM + s] = (bf16_t)val;
                else
                    out[(((size_t)(bb * NUM_HEADS + hId)) * S_DIM + s) * D_K + d] = (bf16_t)val;
            }
        }
    }
}

// ------------------------------------------------------------ flash attention
// Each wave owns 16 query rows (wave-independent: causal early exit is safe).
__global__ __launch_bounds__(256) void attn_kernel(const bf16_t* __restrict__ qb,
                                                   const bf16_t* __restrict__ kb,
                                                   const bf16_t* __restrict__ vtb,
                                                   bf16_t* __restrict__ attnb) {
    __shared__ __align__(16) bf16_t psh[8][16 * 32];   // 1 KB per wave

    const int lane = threadIdx.x & 31;
    const int wid  = threadIdx.x >> 5;
    const int qblk = blockIdx.x & 15;       // S / 128
    const int bh   = blockIdx.x >> 4;       // b * 16 + h
    const int h    = bh & 15, b = bh >> 4;
    const int q0   = qblk * 128 + wid * 16;

    const bf16_t* qbase = qb  + (size_t)bh * S_DIM * D_K;
    const bf16_t* kbase = kb  + (size_t)bh * S_DIM * D_K;
    const bf16_t* vbase = vtb + (size_t)bh * D_K * S_DIM;

    FragBF qa[2];
    load_a_bf16(qa[0], qbase, D_K, q0, 0,  lane);
    load_a_bf16(qa[1], qbase, D_K, q0, 32, lane);

    v8f acc[4];
#pragma unroll
    for (int t = 0; t < 4; ++t)
        acc[t] = (v8f){0.f, 0.f, 0.f, 0.f, 0.f, 0.f, 0.f, 0.f};
    float mrow[8], lrow[8];
#pragma unroll
    for (int i = 0; i < 8; ++i) { mrow[i] = -3e38f; lrow[i] = 0.f; }

    const int hh = lane >> 4;
    const int cn = lane & 15;
    const int qhi = q0 + 15;

    for (int j0 = 0; j0 <= qhi; j0 += 32) {
        // ---- S = Q . K^T  (two N-subtiles of 16 columns, K=64 in 2 steps)
        FragBF kf00, kf01, kf10, kf11;
        load_b_bf16(kf00, kbase, D_K, j0,      0,  lane);
        load_b_bf16(kf01, kbase, D_K, j0,      32, lane);
        load_b_bf16(kf10, kbase, D_K, j0 + 16, 0,  lane);
        load_b_bf16(kf11, kbase, D_K, j0 + 16, 32, lane);

        // V^T fragments are independent of the softmax: issue them early so
        // their latency hides under the softmax VALU work below.
        FragBF vf[4];
#pragma unroll
        for (int t = 0; t < 4; ++t)
            load_b_bf16(vf[t], vbase, S_DIM, t * 16, j0, lane);

        v8f s0 = (v8f){0.f, 0.f, 0.f, 0.f, 0.f, 0.f, 0.f, 0.f};
        v8f s1 = s0;
        s0 = wmma_bf16(qa[0], kf00, s0);
        s0 = wmma_bf16(qa[1], kf01, s0);
        s1 = wmma_bf16(qa[0], kf10, s1);
        s1 = wmma_bf16(qa[1], kf11, s1);

        // prefetch next K / V^T tiles (L2-resident fast path)
        if (j0 + 32 <= qhi) {
            __builtin_prefetch(kbase + (size_t)(j0 + 32 + cn) * D_K, 0, 0);
            __builtin_prefetch(vbase + (size_t)cn * S_DIM + j0 + 32, 0, 0);
        }

        // ---- online softmax over the 32 new columns
        float p0a[8], p1a[8];
#pragma unroll
        for (int i = 0; i < 8; ++i) {
            const int row = q0 + i + 8 * hh;
            float v0 = (j0 + cn      <= row) ? s0[i] * SM_SCALE : NEG_BIG;
            float v1 = (j0 + 16 + cn <= row) ? s1[i] * SM_SCALE : NEG_BIG;
            float rmax = fmaxf(v0, v1);
#pragma unroll
            for (int off = 1; off < 16; off <<= 1)
                rmax = fmaxf(rmax, __shfl_xor(rmax, off, 32));
            const float mnew = fmaxf(mrow[i], rmax);
            const float corr = __expf(mrow[i] - mnew);
            mrow[i] = mnew;
            const float e0 = __expf(v0 - mnew);
            const float e1 = __expf(v1 - mnew);
            float rs = e0 + e1;
#pragma unroll
            for (int off = 1; off < 16; off <<= 1)
                rs += __shfl_xor(rs, off, 32);
            lrow[i] = lrow[i] * corr + rs;
#pragma unroll
            for (int t = 0; t < 4; ++t) acc[t][i] *= corr;
            p0a[i] = e0;
            p1a[i] = e1;
        }

        // ---- transpose P (C layout -> A layout) through wave-private LDS
        bf16_t* ps = psh[wid];
#pragma unroll
        for (int i = 0; i < 8; ++i) {
            const int row = i + 8 * hh;
            ps[row * 32 + cn]      = (bf16_t)p0a[i];
            ps[row * 32 + cn + 16] = (bf16_t)p1a[i];
        }
        asm volatile("s_wait_dscnt 0" ::: "memory");
        FragBF pa;
        {
            const bf16_t* pp = ps + cn * 32 + hh * 8;
            pa.h[0] = *(const v8bf*)(pp);
            pa.h[1] = *(const v8bf*)(pp + 16);
        }

        // ---- O += P . V   (V^T layout -> contiguous B fragments)
#pragma unroll
        for (int t = 0; t < 4; ++t)
            acc[t] = wmma_bf16(pa, vf[t], acc[t]);
    }

    // ---- normalize and store to [b, s, h*64 + d] (bf16)
#pragma unroll
    for (int i = 0; i < 8; ++i) {
        const float inv = 1.0f / lrow[i];
        const int row = q0 + i + 8 * hh;
#pragma unroll
        for (int t = 0; t < 4; ++t) {
            const float val = acc[t][i] * inv;
            attnb[((size_t)b * S_DIM + row) * D_MODEL + h * D_K + t * 16 + cn] = (bf16_t)val;
        }
    }
}

// ------------------------------------------------------------ output GEMM
// out(fp32) = attn(bf16, M x K) @ Wo(bf16, N x K)^T + bo
__global__ __launch_bounds__(256) void outproj_kernel(const bf16_t* __restrict__ A,
                                                      const bf16_t* __restrict__ W,
                                                      const float* __restrict__ bias,
                                                      float* __restrict__ out) {
    const int lane = threadIdx.x & 31;
    const int wid  = threadIdx.x >> 5;
    const int wm = wid & 3, wn = wid >> 2;
    const int gx = blockIdx.x & 7;
    const int gy = blockIdx.x >> 3;
    const int m0 = gy * 128 + wm * 32;
    const int n0 = gx * 128 + wn * 64;

    v8f acc[2][4];
#pragma unroll
    for (int ms = 0; ms < 2; ++ms)
#pragma unroll
        for (int ns = 0; ns < 4; ++ns)
            acc[ms][ns] = (v8f){0.f, 0.f, 0.f, 0.f, 0.f, 0.f, 0.f, 0.f};

    for (int k0 = 0; k0 < D_MODEL; k0 += 32) {
        FragBF a[2];
        load_a_bf16(a[0], A, D_MODEL, m0,      k0, lane);
        load_a_bf16(a[1], A, D_MODEL, m0 + 16, k0, lane);
        FragBF b[4];
#pragma unroll
        for (int ns = 0; ns < 4; ++ns)
            load_b_bf16(b[ns], W, D_MODEL, n0 + ns * 16, k0, lane);
#pragma unroll
        for (int ms = 0; ms < 2; ++ms)
#pragma unroll
            for (int ns = 0; ns < 4; ++ns)
                acc[ms][ns] = wmma_bf16(a[ms], b[ns], acc[ms][ns]);
    }

    const int hh = lane >> 4;
    const int cn = lane & 15;
#pragma unroll
    for (int ms = 0; ms < 2; ++ms) {
#pragma unroll
        for (int ns = 0; ns < 4; ++ns) {
            const int n = n0 + ns * 16 + cn;
            const float bv = bias[n];
#pragma unroll
            for (int i = 0; i < 8; ++i) {
                const int row = m0 + ms * 16 + i + 8 * hh;
                out[(size_t)row * D_MODEL + n] = acc[ms][ns][i] + bv;
            }
        }
    }
}

// ---------------------------------------------------------------- launcher
extern "C" void kernel_launch(void* const* d_in, const int* in_sizes, int n_in,
                              void* d_out, int out_size, void* d_ws, size_t ws_size,
                              hipStream_t stream) {
    (void)in_sizes; (void)n_in; (void)out_size; (void)ws_size;

    const float* Q  = (const float*)d_in[0];
    const float* K  = (const float*)d_in[1];
    const float* V  = (const float*)d_in[2];
    // d_in[3] is the causal mask; applied analytically in attn_kernel.
    const float* Wq = (const float*)d_in[4];
    const float* bq = (const float*)d_in[5];
    const float* Wk = (const float*)d_in[6];
    const float* bk = (const float*)d_in[7];
    const float* Wv = (const float*)d_in[8];
    const float* bv = (const float*)d_in[9];
    const float* Wo = (const float*)d_in[10];
    const float* bo = (const float*)d_in[11];

    char* ws = (char*)d_ws;
    const size_t WBYTES = (size_t)D_MODEL * D_MODEL * sizeof(bf16_t);   // 2 MB
    const size_t ABYTES = (size_t)M_ROWS * D_MODEL * sizeof(bf16_t);    // 16 MB
    bf16_t* Wqb = (bf16_t*)(ws + 0 * WBYTES);
    bf16_t* Wkb = (bf16_t*)(ws + 1 * WBYTES);
    bf16_t* Wvb = (bf16_t*)(ws + 2 * WBYTES);
    bf16_t* Wob = (bf16_t*)(ws + 3 * WBYTES);
    size_t off = 4 * WBYTES;
    bf16_t* q_b  = (bf16_t*)(ws + off); off += ABYTES;   // [B,H,S,Dk]
    bf16_t* k_b  = (bf16_t*)(ws + off); off += ABYTES;   // [B,H,S,Dk]
    bf16_t* vt_b = (bf16_t*)(ws + off); off += ABYTES;   // [B,H,Dk,S]
    bf16_t* at_b = (bf16_t*)(ws + off);                  // [B,S,D]

    const int nW = D_MODEL * D_MODEL;
    cvt_f32_bf16<<<4096, 256, 0, stream>>>(Wq, Wqb, nW);
    cvt_f32_bf16<<<4096, 256, 0, stream>>>(Wk, Wkb, nW);
    cvt_f32_bf16<<<4096, 256, 0, stream>>>(Wv, Wvb, nW);
    cvt_f32_bf16<<<4096, 256, 0, stream>>>(Wo, Wob, nW);

    // (M/128)*(N/128) = 64 * 8 = 512 blocks
    proj_kernel<0><<<512, 256, 0, stream>>>(Q, Wqb, bq, q_b);
    proj_kernel<0><<<512, 256, 0, stream>>>(K, Wkb, bk, k_b);
    proj_kernel<1><<<512, 256, 0, stream>>>(V, Wvb, bv, vt_b);

    // B*H*(S/128) = 4*16*16 = 1024 blocks
    attn_kernel<<<1024, 256, 0, stream>>>(q_b, k_b, vt_b, at_b);

    outproj_kernel<<<512, 256, 0, stream>>>(at_b, Wob, bo, (float*)d_out);
}
